// CustomEncoder_36386962932318
// MI455X (gfx1250) — matrix operand order repaired and checked by reading, not compiled, not run on hardware
//
#include <hip/hip_runtime.h>

// ---------------------------------------------------------------------------
// MI455X / gfx1250 implementation of the VMamba-style encoder.
// Dense GEMMs use v_wmma_f32_16x16x32_bf16 (bf16 in, f32 accumulate).
// GEMM: 128-thread block (4 waves) computes a 32x32 tile; LDS staged with
// ds_store_b128, fragments gathered with ds_load_b128 (A row-major, B
// transposed [n][k]); full-tile fast path avoids per-lane bounds branches;
// k-unit-stride operands load via global_load_b128 (uniform runtime check).
// SSM recurrences: lane-parallel serial scans (parallel over b x ch; state unrolled per lane).
// Workspace: ~188 MB carved from d_ws as fixed slots.
// ---------------------------------------------------------------------------

typedef __attribute__((ext_vector_type(16))) __bf16          v16bf;
typedef __attribute__((ext_vector_type(8)))  float           v8f;
typedef __attribute__((ext_vector_type(4)))  float           f32x4;
typedef __attribute__((ext_vector_type(4)))  unsigned int    u32x4;
typedef __attribute__((ext_vector_type(8)))  unsigned int    u32x8;

#define EW_BLK 256
static inline dim3 NB(long n) { return dim3((unsigned)((n + EW_BLK - 1) / EW_BLK)); }

__device__ __forceinline__ unsigned short f2bf(float f) {
  unsigned int u = __float_as_uint(f);
  u += 0x7fffu + ((u >> 16) & 1u);          // round-to-nearest-even
  return (unsigned short)(u >> 16);
}
__device__ __forceinline__ unsigned int pack2bf(float a, float b) {
  return (unsigned int)f2bf(a) | ((unsigned int)f2bf(b) << 16);
}
__device__ __forceinline__ float siluf(float x) { return x / (1.f + __expf(-x)); }

// ---------------------------------------------------------------------------
// Generic strided batched GEMM.
//   C[b,m,n] = act( sum_k A[b,m,k]*B[b,k,n] + bias[m*bsm + n*bsn] )
// Element A + b*ab + m*am + k*ak (same scheme for B, C). act: 0=none, 1=silu.
// Block = 128 threads = 4 waves; block tile 32x32, one 16x16 WMMA quadrant
// per wave; K-step 32.
// LDS pitch 40 halfs (80 B) keeps every 8-half fragment run 16 B aligned.
// ---------------------------------------------------------------------------
#define TPITCH 40

__global__ void __launch_bounds__(128)
k_gemm(const float* __restrict__ A, long am, long ak, long ab,
       const float* __restrict__ B, long bk, long bn, long bb,
       float* __restrict__ C, long cm, long cn, long cb,
       const float* __restrict__ bias, int bsm, int bsn,
       int M, int N, int K, int act) {
  __shared__ __align__(16) unsigned short lA[32][TPITCH];    // [m][k]
  __shared__ __align__(16) unsigned short lBt[32][TPITCH];   // [n][k] (transposed)
  const int bz = blockIdx.z;
  const int m0 = blockIdx.y * 32, n0 = blockIdx.x * 32;
  const float* Ab = A + (long)bz * ab;
  const float* Bb = B + (long)bz * bb;
  const int t = threadIdx.x;
  const int w = t >> 5, l = t & 31;          // wave id, lane id (wave32)
  const int wm = (w >> 1) * 16, wn = (w & 1) * 16;
  const int rowA = t >> 2;                   // 0..31: m-row for A, n-row for B
  const int kj0 = (t & 3) * 8;               // 8-wide k sub-run per thread
  // Block-uniform: operand is k-unit-stride and every thread's 8-run is 16B
  // aligned (row stride multiple of 4 floats, 16B-aligned base).
  const bool va = (ak == 1) && ((am & 3) == 0) && ((((unsigned long)Ab) & 15) == 0);
  const bool vb = (bk == 1) && ((bn & 3) == 0) && ((((unsigned long)Bb) & 15) == 0);
  v8f acc = {};
  for (int k0 = 0; k0 < K; k0 += 32) {
    float av[8], bv[8];
    if ((m0 + 32 <= M) && (n0 + 32 <= N) && (k0 + 32 <= K)) {
      // full tile: straight-line unguarded loads
      const float* ap = Ab + (long)(m0 + rowA) * am + (long)(k0 + kj0) * ak;
      const float* bp = Bb + (long)(k0 + kj0) * bk + (long)(n0 + rowA) * bn;
      if (va) {                              // global_load_b128 x2
        f32x4 q0 = ((const f32x4*)ap)[0];
        f32x4 q1 = ((const f32x4*)ap)[1];
#pragma unroll
        for (int e = 0; e < 4; ++e) { av[e] = q0[e]; av[e + 4] = q1[e]; }
      } else {
#pragma unroll
        for (int e = 0; e < 8; ++e) av[e] = ap[(long)e * ak];
      }
      if (vb) {                              // global_load_b128 x2
        f32x4 q0 = ((const f32x4*)bp)[0];
        f32x4 q1 = ((const f32x4*)bp)[1];
#pragma unroll
        for (int e = 0; e < 4; ++e) { bv[e] = q0[e]; bv[e + 4] = q1[e]; }
      } else {
#pragma unroll
        for (int e = 0; e < 8; ++e) bv[e] = bp[(long)e * bk];
      }
      if (k0 + 64 <= K) {                    // global_prefetch_b8 next K tile
        __builtin_prefetch(ap + 32 * ak, 0, 1);
        __builtin_prefetch(bp + 32 * bk, 0, 1);
      }
    } else {
#pragma unroll
      for (int e = 0; e < 8; ++e) {
        int kk = k0 + kj0 + e;
        av[e] = ((m0 + rowA) < M && kk < K)
                    ? Ab[(long)(m0 + rowA) * am + (long)kk * ak] : 0.f;
        bv[e] = ((n0 + rowA) < N && kk < K)
                    ? Bb[(long)kk * bk + (long)(n0 + rowA) * bn] : 0.f;
      }
    }
    u32x4 pa, pb;
#pragma unroll
    for (int e = 0; e < 4; ++e) {
      pa[e] = pack2bf(av[2 * e], av[2 * e + 1]);
      pb[e] = pack2bf(bv[2 * e], bv[2 * e + 1]);
    }
    *(u32x4*)&lA[rowA][kj0]  = pa;           // ds_store_b128
    *(u32x4*)&lBt[rowA][kj0] = pb;           // ds_store_b128
    __syncthreads();
    // Fragment gather per CDNA5 16-bit WMMA VGPR layouts (ISA 7.12.2):
    // A lane: row l&15 (+quadrant), halfs [kb..kb+7] and [kb+16..kb+23], kb=8*(l>=16)
    // B lane: col l&15 (+quadrant), halfs [kb2..kb2+7] and [kb2+16..kb2+23], kb2=8*(l>=16)
    u32x8 ua, ub;
    {
      const int mr = wm + (l & 15), kb = (l >> 4) * 8;
      u32x4 c0 = *(const u32x4*)&lA[mr][kb];        // ds_load_b128
      u32x4 c1 = *(const u32x4*)&lA[mr][kb + 16];   // ds_load_b128
#pragma unroll
      for (int e = 0; e < 4; ++e) { ua[e] = c0[e]; ua[e + 4] = c1[e]; }
    }
    {
      const int nc = wn + (l & 15), kb2 = (l >> 4) * 8;
      u32x4 c0 = *(const u32x4*)&lBt[nc][kb2];      // ds_load_b128
      u32x4 c1 = *(const u32x4*)&lBt[nc][kb2 + 16]; // ds_load_b128
#pragma unroll
      for (int e = 0; e < 4; ++e) { ub[e] = c0[e]; ub[e + 4] = c1[e]; }
    }
    v16bf af = __builtin_bit_cast(v16bf, ua);
    v16bf bf = __builtin_bit_cast(v16bf, ub);
    acc = __builtin_amdgcn_wmma_f32_16x16x32_bf16(false, af, false, bf, (short)0, acc,
                                                  false, false);
    __syncthreads();
  }
  const int nc = l & 15;
#pragma unroll
  for (int r = 0; r < 8; ++r) {              // D layout: M = r + 8*(lane>=16)
    int gm = m0 + wm + r + ((l >> 4) << 3);
    int gn = n0 + wn + nc;
    if (gm < M && gn < N) {
      float v = acc[r];
      if (bias) v += bias[gm * bsm + gn * bsn];
      if (act == 1) v = siluf(v);
      C[(long)bz * cb + (long)gm * cm + (long)gn * cn] = v;
    }
  }
}

// ---------------------------------------------------------------------------
// Patch embed: 2x2/stride2 conv (3->64) + bias + 2D sinusoidal pos-enc.
// ---------------------------------------------------------------------------
__global__ void k_patch(const float* __restrict__ in, const float* __restrict__ w,
                        const float* __restrict__ bias, float* __restrict__ out,
                        int Bn, int Ci, int Hi, int Wi, int Co, int Ho, int Wo) {
  long n = (long)Bn * Co * Ho * Wo;
  long t = (long)blockIdx.x * blockDim.x + threadIdx.x;
  if (t >= n) return;
  int x = t % Wo; long r = t / Wo;
  int y = r % Ho; r /= Ho;
  int o = r % Co; int b = r / Co;
  float s = bias[o];
  for (int i = 0; i < Ci; ++i)
#pragma unroll
    for (int ky = 0; ky < 2; ++ky)
#pragma unroll
      for (int kx = 0; kx < 2; ++kx)
        s += in[((long)(b * Ci + i) * Hi + 2 * y + ky) * Wi + 2 * x + kx] *
             w[((o * Ci + i) * 2 + ky) * 2 + kx];
  int dm = Co >> 1;
  int oo = (o < dm) ? o : o - dm;
  int k = oo >> 1;
  float dv = __expf(-(float)(2 * k) * 9.210340371976184f / (float)dm); // ln(1e4)
  float pos = (o < dm) ? (float)x : (float)y;
  float a = pos * dv;
  s += ((oo & 1) == 0) ? sinf(a) : cosf(a);
  out[t] = s;
}

// Depthwise causal conv1d (k=4) + SiLU over xz[..., :di] -> u
__global__ void k_conv1d(const float* __restrict__ xz, const float* __restrict__ w,
                         const float* __restrict__ bias, float* __restrict__ u,
                         int Bn, int L, int di) {
  long n = (long)Bn * L * di;
  long t = (long)blockIdx.x * blockDim.x + threadIdx.x;
  if (t >= n) return;
  int c = t % di; long bl = t / di;
  int l = bl % L; int b = bl / L;
  const float* xb = xz + (long)b * L * 2 * di;
  float s = bias[c];
#pragma unroll
  for (int j = 0; j < 4; ++j) {
    int tt = l + j - 3;
    if (tt >= 0) s += w[c * 4 + j] * xb[(long)tt * 2 * di + c];
  }
  u[t] = siluf(s);
}

// dbl[row, r] = sum_c u[row,c] * xproj_w[r, c]
__global__ void k_xproj(const float* __restrict__ u, const float* __restrict__ xw,
                        float* __restrict__ dbl, int Bn, int L, int di, int nd) {
  long n = (long)Bn * L * nd;
  long t = (long)blockIdx.x * blockDim.x + threadIdx.x;
  if (t >= n) return;
  int rr = t % nd; long row = t / nd;
  const float* up = u + row * di;
  const float* wp = xw + (long)rr * di;
  float s = 0.f;
  for (int c = 0; c < di; ++c) s += up[c] * wp[c];
  dbl[row * nd + rr] = s;
}

// dt[row, c] = softplus( dbl[row,:dtr] . dt_w[c,:] + dt_b[c] )
__global__ void k_dt(const float* __restrict__ dbl, const float* __restrict__ dw,
                     const float* __restrict__ db, float* __restrict__ dt,
                     int Bn, int L, int di, int dtr, int nd) {
  long n = (long)Bn * L * di;
  long t = (long)blockIdx.x * blockDim.x + threadIdx.x;
  if (t >= n) return;
  int c = t % di; long row = t / di;
  const float* dp = dbl + row * nd;
  float s = db[c];
  for (int r = 0; r < dtr; ++r) s += dp[r] * dw[c * dtr + r];
  dt[t] = (s > 20.f) ? s : log1pf(__expf(s));
}

// Selective scan: serial over L, parallel over (b, c); D_STATE = 2 unrolled.
// ym[row,c] = (y + u*D) * silu(z)
__global__ void k_scan(const float* __restrict__ dt, const float* __restrict__ u,
                       const float* __restrict__ dbl, const float* __restrict__ xz,
                       const float* __restrict__ A_log, const float* __restrict__ Dp,
                       float* __restrict__ ym, int Bn, int L, int di, int dtr, int nd) {
  int t = blockIdx.x * blockDim.x + threadIdx.x;
  if (t >= Bn * di) return;
  int c = t % di, b = t / di;
  float a0 = -__expf(A_log[c * 2 + 0]);
  float a1 = -__expf(A_log[c * 2 + 1]);
  float Dv = Dp[c];
  float h0 = 0.f, h1 = 0.f;
  for (int l = 0; l < L; ++l) {
    long row = (long)b * L + l;
    float dtv = dt[row * di + c];
    float uv = u[row * di + c];
    const float* dp = dbl + row * nd;
    float B0 = dp[dtr], B1 = dp[dtr + 1], C0 = dp[dtr + 2], C1 = dp[dtr + 3];
    float du = dtv * uv;
    h0 = __expf(dtv * a0) * h0 + du * B0;
    h1 = __expf(dtv * a1) * h1 + du * B1;
    float y = h0 * C0 + h1 * C1 + uv * Dv;
    float zv = xz[row * 2 * di + di + c];
    ym[row * di + c] = y * siluf(zv);
  }
}

// LayerNorm over width O of row-major g2, strided writeback:
// out[b*O*L + o*fs + l*ts]
__global__ void k_ln(const float* __restrict__ g2, const float* __restrict__ g,
                     const float* __restrict__ beta, float* __restrict__ out,
                     int Bn, int L, int O, long fs, long ts) {
  long row = (long)blockIdx.x * blockDim.x + threadIdx.x;
  if (row >= (long)Bn * L) return;
  int b = row / L, l = row % L;
  const float* x = g2 + row * O;
  float m = 0.f;
  for (int o = 0; o < O; ++o) m += x[o];
  m /= O;
  float v = 0.f;
  for (int o = 0; o < O; ++o) { float d = x[o] - m; v += d * d; }
  v /= O;
  float inv = rsqrtf(v + 1e-5f);
  float* ob = out + (long)b * O * L;
  for (int o = 0; o < O; ++o)
    ob[(long)o * fs + (long)l * ts] = (x[o] - m) * inv * g[o] + beta[o];
}

__global__ void k_flip(const float* __restrict__ in, float* __restrict__ out,
                       int Bn, int C, int H, int W) {
  long n = (long)Bn * C * H * W;
  long t = (long)blockIdx.x * blockDim.x + threadIdx.x;
  if (t >= n) return;
  int w = t % W; long r = t / W;
  int h = r % H; long bc = r / H;
  out[(bc * H + (H - 1 - h)) * W + (W - 1 - w)] = in[t];
}

__global__ void k_pool(const float* __restrict__ in, float* __restrict__ out,
                       int Bn, int C, int H, int W) {
  int Ho = H >> 1, Wo = W >> 1;
  long n = (long)Bn * C * Ho * Wo;
  long t = (long)blockIdx.x * blockDim.x + threadIdx.x;
  if (t >= n) return;
  int wo = t % Wo; long r = t / Wo;
  int ho = r % Ho; long bc = r / Ho;
  const float* p = in + (bc * H + 2 * ho) * W + 2 * wo;
  out[t] = 0.25f * (p[0] + p[1] + p[W] + p[W + 1]);
}

__global__ void k_add(const float* a, const float* b, float* o, long n) {
  long t = (long)blockIdx.x * blockDim.x + threadIdx.x;
  if (t < n) o[t] = a[t] + b[t];
}
__global__ void k_addip(float* a, const float* b, long n) {
  long t = (long)blockIdx.x * blockDim.x + threadIdx.x;
  if (t < n) a[t] += b[t];
}
__global__ void k_addsilu(const float* a, const float* b, float* o, long n) {
  long t = (long)blockIdx.x * blockDim.x + threadIdx.x;
  if (t < n) o[t] = siluf(a[t] + b[t]);
}

__global__ void k_mean(const float* __restrict__ in, float* __restrict__ out,
                       int Bn, int C, int HW) {
  int t = blockIdx.x * blockDim.x + threadIdx.x;
  if (t >= Bn * C) return;
  const float* p = in + (long)t * HW;
  float s = 0.f;
  for (int i = 0; i < HW; ++i) s += p[i];
  out[t] = s / (float)HW;
}

// act: 0=none, 2=hardswish (x * clip(x+3,0,6)/6)
__global__ void k_fc(const float* __restrict__ in, const float* __restrict__ w,
                     const float* __restrict__ bias, float* __restrict__ out,
                     int Bn, int K, int N, int act) {
  int t = blockIdx.x * blockDim.x + threadIdx.x;
  if (t >= Bn * N) return;
  int j = t % N, b = t / N;
  const float* ip = in + (long)b * K;
  const float* wp = w + (long)j * K;
  float s = bias[j];
  for (int k = 0; k < K; ++k) s += ip[k] * wp[k];
  if (act == 2) {
    float c = s + 3.f;
    c = fminf(fmaxf(c, 0.f), 6.f);
    s = s * c * (1.f / 6.f);
  }
  out[t] = s;
}

// ---------------------------------------------------------------------------
// Host-side orchestration
// ---------------------------------------------------------------------------
namespace {

struct MambaP {
  const float *in_w, *conv_w, *conv_b, *xproj_w, *dt_w, *dt_b, *A_log, *D, *out_w;
  int dm, oc, dtr;
};
struct ChaP { MambaP m; const float *g, *beta; };
struct LinP { const float *w, *b; };
struct BrP {
  LinP lin; ChaP ssm1; LinP lin2; ChaP ssm2;
  int has3; ChaP spa; LinP lin3; const float *sp_w, *sp_b; LinP sp_lc;
};

struct Reader {
  void* const* d; int i;
  const float* next() { return (const float*)d[i++]; }
};
MambaP rd_mamba(Reader& r, int dm, int oc) {
  MambaP p;
  p.in_w = r.next(); p.conv_w = r.next(); p.conv_b = r.next();
  p.xproj_w = r.next(); p.dt_w = r.next(); p.dt_b = r.next();
  p.A_log = r.next(); p.D = r.next(); p.out_w = r.next();
  p.dm = dm; p.oc = oc; p.dtr = (dm + 15) / 16;
  return p;
}
ChaP rd_cha(Reader& r, int c, int o) {
  ChaP x; x.m = rd_mamba(r, c, o); x.g = r.next(); x.beta = r.next(); return x;
}
LinP rd_lin(Reader& r) { LinP l; l.w = r.next(); l.b = r.next(); return l; }
BrP rd_br12(Reader& r, int ci, int co) {
  BrP b{};
  b.lin = rd_lin(r); b.ssm1 = rd_cha(r, ci, ci);
  b.lin2 = rd_lin(r); b.ssm2 = rd_cha(r, co, co);
  b.has3 = 0;
  return b;
}
BrP rd_br123(Reader& r, int ci, int co, int sp) {
  BrP b = rd_br12(r, ci, co);
  b.has3 = 1;
  b.spa = rd_cha(r, sp * sp, sp * sp);
  b.lin3 = rd_lin(r);
  b.sp_w = r.next(); b.sp_b = r.next();
  b.sp_lc = rd_lin(r);
  return b;
}

struct WS {
  float *Y, *P, *A1, *A2, *A3, *A4, *A5, *A6;   // BIG activation slots
  float *U, *DT, *YM, *G2, *XZ, *DBL, *G, *H1;  // mamba scratch + head
};

void gemm(hipStream_t st, const float* A, long am, long ak, long ab,
          const float* B, long bk, long bn, long bb,
          float* C, long cm, long cn, long cb,
          const float* bias, int bsm, int bsn, int M, int N, int K, int act) {
  dim3 g((N + 31) / 32, (M + 31) / 32, 8);
  k_gemm<<<g, dim3(128), 0, st>>>(A, am, ak, ab, B, bk, bn, bb, C, cm, cn, cb,
                                  bias, bsm, bsn, M, N, K, act);
}

// out[b,o,l] = silu( sum_c w[o,c]*X[b,c,l] + b[o] )
void linlayer(hipStream_t st, const LinP& ln, const float* X, float* out,
              int Ci, int Co, int L) {
  gemm(st, ln.w, Ci, 1, 0, X, L, 1, (long)Ci * L, out, L, 1, (long)Co * L,
       ln.b, 1, 0, Co, L, Ci, 1);
}

// Full Mamba block + LayerNorm. X token row m, feature k at X + b*ab + m*am + k*ak.
// LN writeback: out[b*oc*L + o*fs + l*ts].
void run_mamba(hipStream_t st, const ChaP& ch, const float* X, long am, long ak, long ab,
               float* out, long fs, long ts, int L, WS& w) {
  const MambaP& p = ch.m;
  const int Bn = 8, dm = p.dm, oc = p.oc, dtr = p.dtr, nd = dtr + 4;
  // xz = x @ in_w.T   (L x 2dm)
  gemm(st, X, am, ak, ab, p.in_w, 1, dm, 0, w.XZ, 2 * dm, 1, (long)L * 2 * dm,
       nullptr, 0, 0, L, 2 * dm, dm, 0);
  long n1 = (long)Bn * L * dm;
  k_conv1d<<<NB(n1), EW_BLK, 0, st>>>(w.XZ, p.conv_w, p.conv_b, w.U, Bn, L, dm);
  long n2 = (long)Bn * L * nd;
  k_xproj<<<NB(n2), EW_BLK, 0, st>>>(w.U, p.xproj_w, w.DBL, Bn, L, dm, nd);
  k_dt<<<NB(n1), EW_BLK, 0, st>>>(w.DBL, p.dt_w, p.dt_b, w.DT, Bn, L, dm, dtr, nd);
  k_scan<<<NB((long)Bn * dm), EW_BLK, 0, st>>>(w.DT, w.U, w.DBL, w.XZ, p.A_log, p.D,
                                               w.YM, Bn, L, dm, dtr, nd);
  // g2 = ym @ out_w.T  (L x oc)
  gemm(st, w.YM, dm, 1, (long)L * dm, p.out_w, 1, dm, 0, w.G2, oc, 1, (long)L * oc,
       nullptr, 0, 0, L, oc, dm, 0);
  k_ln<<<NB((long)Bn * L), EW_BLK, 0, st>>>(w.G2, ch.g, ch.beta, out, Bn, L, oc, fs, ts);
}

void run_branch(hipStream_t st, const BrP& br, const float* X, float* OUT,
                int ci, int co, int H, int Wd, WS& w) {
  const int L = H * Wd;
  const long nco = 8L * co * L;
  // x1 = linlayer(lin, X)                                -> A4 (accumulator)
  linlayer(st, br.lin, X, w.A4, ci, co, L);
  // x2 = ssm_cha(ssm2, linlayer(lin2, ssm_cha(ssm1, X)))
  run_mamba(st, br.ssm1, X, 1, L, (long)ci * L, w.A5, L, 1, L, w);   // channel SSM
  linlayer(st, br.lin2, w.A5, w.A6, ci, co, L);
  run_mamba(st, br.ssm2, w.A6, 1, L, (long)co * L, w.A5, L, 1, L, w);
  if (!br.has3) {
    k_add<<<NB(nco), EW_BLK, 0, st>>>(w.A4, w.A5, OUT, nco);
    return;
  }
  k_addip<<<NB(nco), EW_BLK, 0, st>>>(w.A4, w.A5, nco);
  // x3 = linlayer(lin3, ssm_spa(spa, X)); spa: tokens = channels, features = hw
  const int hw = L;
  run_mamba(st, br.spa, X, hw, 1, (long)ci * hw, w.A5, 1, hw, ci, w);
  linlayer(st, br.lin3, w.A5, w.A6, ci, co, L);
  k_addip<<<NB(nco), EW_BLK, 0, st>>>(w.A4, w.A6, nco);
  // xr = linlayer(lc, silu(X @ sp_w.T + sp_b))  (per-channel spatial FC)
  gemm(st, X, hw, 1, (long)ci * hw, br.sp_w, 1, hw, 0, w.A5, hw, 1, (long)ci * hw,
       br.sp_b, 0, 1, ci, hw, hw, 1);
  linlayer(st, br.sp_lc, w.A5, w.A6, ci, co, L);
  k_add<<<NB(nco), EW_BLK, 0, st>>>(w.A4, w.A6, OUT, nco);
}

// stage(fn, t) = silu( fn(t) + flip(fn(flip(t))) )
void run_stage(hipStream_t st, const BrP& br, const float* Xin, float* Yout,
               int ci, int co, int H, int Wd, WS& w) {
  const long nin = 8L * ci * H * Wd, nout = 8L * co * H * Wd;
  k_flip<<<NB(nin), EW_BLK, 0, st>>>(Xin, w.A1, 8, ci, H, Wd);
  run_branch(st, br, Xin, w.A2, ci, co, H, Wd, w);
  run_branch(st, br, w.A1, w.A3, ci, co, H, Wd, w);
  k_flip<<<NB(nout), EW_BLK, 0, st>>>(w.A3, w.A1, 8, co, H, Wd);
  k_addsilu<<<NB(nout), EW_BLK, 0, st>>>(w.A2, w.A1, Yout, nout);
}

} // namespace

extern "C" void kernel_launch(void* const* d_in, const int* in_sizes, int n_in,
                              void* d_out, int out_size, void* d_ws, size_t ws_size,
                              hipStream_t stream) {
  (void)in_sizes; (void)out_size; (void)ws_size;
  if (n_in < 171) return;   // expected 171 flat arrays in dict-DFS order

  Reader r{d_in, 0};
  const float* inp  = r.next();
  const float* pm_w = r.next();
  const float* pm_b = r.next();
  BrP inc = rd_br12(r, 64, 64);
  BrP d1  = rd_br12(r, 64, 128);
  BrP d2  = rd_br12(r, 128, 256);
  BrP d3  = rd_br123(r, 256, 512, 10);
  BrP d4  = rd_br123(r, 512, 1024, 5);
  const float* w1 = r.next(); const float* b1 = r.next();
  const float* w2 = r.next(); const float* b2 = r.next();

  // Workspace slots (floats). BIG covers the stage-1 maxima.
  const size_t BIG = 3276800;   // 8 * 64 * 80 * 80
  const size_t XZN = 6553600;   // 8 * 6400 * 128
  const size_t DBN = 1048576;
  float* base = (float*)d_ws;
  size_t off = 0;
  auto take = [&](size_t n) { float* p = base + off; off += n; return p; };
  WS w;
  w.Y  = take(BIG); w.P  = take(BIG); w.A1 = take(BIG); w.A2 = take(BIG);
  w.A3 = take(BIG); w.A4 = take(BIG); w.A5 = take(BIG); w.A6 = take(BIG);
  w.U  = take(BIG); w.DT = take(BIG); w.YM = take(BIG); w.G2 = take(BIG);
  w.XZ = take(XZN); w.DBL = take(DBN);
  w.G  = take(8 * 1024); w.H1 = take(8 * 1280);

  // Patch embed + positional encoding -> Y (8,64,80,80)
  k_patch<<<NB(8L * 64 * 80 * 80), EW_BLK, 0, stream>>>(inp, pm_w, pm_b, w.Y,
                                                        8, 3, 160, 160, 64, 80, 80);
  // Stage 1 (in-place on Y)
  run_stage(stream, inc, w.Y, w.Y, 64, 64, 80, 80, w);
  // Stage 2..5: pool then bidirectional branch
  k_pool<<<NB(8L * 64 * 40 * 40), EW_BLK, 0, stream>>>(w.Y, w.P, 8, 64, 80, 80);
  run_stage(stream, d1, w.P, w.Y, 64, 128, 40, 40, w);
  k_pool<<<NB(8L * 128 * 20 * 20), EW_BLK, 0, stream>>>(w.Y, w.P, 8, 128, 40, 40);
  run_stage(stream, d2, w.P, w.Y, 128, 256, 20, 20, w);
  k_pool<<<NB(8L * 256 * 10 * 10), EW_BLK, 0, stream>>>(w.Y, w.P, 8, 256, 20, 20);
  run_stage(stream, d3, w.P, w.Y, 256, 512, 10, 10, w);
  k_pool<<<NB(8L * 512 * 5 * 5), EW_BLK, 0, stream>>>(w.Y, w.P, 8, 512, 10, 10);
  run_stage(stream, d4, w.P, w.Y, 512, 1024, 5, 5, w);

  // Head: global mean pool -> FC(1024->1280, hardswish) -> FC(1280->1000)
  k_mean<<<NB(8L * 1024), EW_BLK, 0, stream>>>(w.Y, w.G, 8, 1024, 25);
  k_fc<<<NB(8L * 1280), EW_BLK, 0, stream>>>(w.G, w1, b1, w.H1, 8, 1024, 1280, 2);
  k_fc<<<NB(8L * 1000), EW_BLK, 0, stream>>>(w.H1, w2, b2, (float*)d_out, 8, 1280, 1000, 0);
}
